// GraphEdgeAttr_47493748359689
// MI455X (gfx1250) — matrix operand order repaired and checked by reading, not compiled, not run on hardware
//
#include <hip/hip_runtime.h>
#include <hip/hip_bf16.h>

#define HD   128          // hidden size
#define NT   8            // N tiles (128/16)
#define NK   4            // K steps for K=128 (128/32)
#define NK1  12           // K steps for K=384
#define STG_STRIDE 136    // staging row stride in halves (128 + 8 pad)

typedef __attribute__((ext_vector_type(16))) __bf16 v16bf;
typedef __attribute__((ext_vector_type(8)))  float  v8f;

union Frag {
    v16bf   v;
    unsigned u[8];
    uint4    q[2];
};

// LDS layout (bytes)
#define WP_OFF   0u            // 8*4 tiles * 1024B  = 32768
#define WC_OFF   32768u
#define W1_OFF   65536u        // 8*12 tiles * 1024B = 98304
#define W2_OFF   163840u
#define BIAS_OFF 196608u       // 4*128 f32 = 2048
#define STG_OFF  198656u       // 8 waves * 32*136*2 = 69632
#define SMEM_BYTES (STG_OFF + 8u * 32u * STG_STRIDE * 2u)   // 268288

// One v_cvt_pk_bf16_f32: packs bf16(a)|bf16(b)<<16 in a single VALU op.
__device__ __forceinline__ unsigned pk2(float a, float b) {
    unsigned r;
    asm("v_cvt_pk_bf16_f32 %0, %1, %2" : "=v"(r) : "v"(a), "v"(b));
    return r;
}
// |bf16(a)|,|bf16(b)| packed: bf16 RNE is sign-symmetric so this equals bf16(|a|),bf16(|b|).
__device__ __forceinline__ unsigned pk2_abs(float a, float b) {
    return pk2(a, b) & 0x7FFF7FFFu;
}

// Load one 32x16 bf16 B fragment from pre-swizzled LDS (tile = nt*nK + ks).
__device__ __forceinline__ Frag load_b(const unsigned char* base, int tile, int lane) {
    Frag f;
    const uint4* p = (const uint4*)(base + ((unsigned)tile * 32u + (unsigned)lane) * 32u);
    f.q[0] = p[0];
    f.q[1] = p[1];
    return f;
}

// Convert f32 weight matrix [N][K] (row-major) to bf16, pre-swizzled per-tile
// into B-fragment order: chunk((nt*nK+ks), lane) = 16 contiguous bf16.
__device__ void stage_w(unsigned char* dst, const float* __restrict__ src, int N, int K) {
    int nK = K >> 5;
    int nchunks = (N >> 4) * nK * 32;
    for (int c = threadIdx.x; c < nchunks; c += blockDim.x) {
        int lane = c & 31;
        int rest = c >> 5;                 // nt*nK + ks
        int ks   = rest % nK;
        int nt   = rest / nK;
        int n    = nt * 16 + (lane & 15);
        int k0   = ks * 32 + ((lane >> 4) << 4);
        const float* s = src + (size_t)n * K + k0;
        unsigned u[8];
        #pragma unroll
        for (int j = 0; j < 8; ++j) u[j] = pk2(s[2 * j], s[2 * j + 1]);
        uint4* d = (uint4*)(dst + ((unsigned)rest * 32u + (unsigned)lane) * 32u);
        d[0] = make_uint4(u[0], u[1], u[2], u[3]);
        d[1] = make_uint4(u[4], u[5], u[6], u[7]);
    }
}

// Read 4 A-fragments (16x32 bf16, K=0..127) from staging rows [rowbase..rowbase+15].
__device__ __forceinline__ void read_stage_A(const unsigned short* stgRow, int lrow, int hl, Frag* f) {
    #pragma unroll
    for (int ks = 0; ks < NK; ++ks) {
        #pragma unroll
        for (int run = 0; run < 2; ++run) {
            int k0 = ks * 32 + 16 * hl + run * 8;
            f[ks].q[run] = *(const uint4*)(stgRow + lrow * STG_STRIDE + k0);
        }
    }
}

// Convert activated accumulator to bf16 (paired) and scatter into staging rows.
__device__ __forceinline__ void store_acc_bf16(unsigned short* stgRow, int col, int hl,
                                               const float* va) {
    #pragma unroll
    for (int r = 0; r < 8; r += 2) {
        unsigned u = pk2(va[r], va[r + 1]);
        int base = (r + 8 * hl) * STG_STRIDE + col;
        stgRow[base]              = (unsigned short)u;
        stgRow[base + STG_STRIDE] = (unsigned short)(u >> 16);
    }
}

// Gather one 16-edge A-tile (row erow per lane) into bf16 A-fragments + |diff|.
__device__ __forceinline__ void gather_frags(const float* __restrict__ x_src,
                                             const float* __restrict__ x_dst,
                                             const int* __restrict__ ei, int E,
                                             int erow, int hl,
                                             Frag* pA, Frag* cA, Frag* dA) {
    int np = ei[erow];
    int nc = ei[E + erow];
    const float* pr = x_src + (size_t)np * HD;
    const float* cr = x_dst + (size_t)nc * HD;
    #pragma unroll
    for (int ks = 0; ks < NK; ++ks) {
        #pragma unroll
        for (int run = 0; run < 2; ++run) {
            int k0 = ks * 32 + 16 * hl + run * 8;
            float4 p0 = *(const float4*)(pr + k0);
            float4 p1 = *(const float4*)(pr + k0 + 4);
            float4 c0 = *(const float4*)(cr + k0);
            float4 c1 = *(const float4*)(cr + k0 + 4);
            int j = run * 4;
            pA[ks].u[j + 0] = pk2(p0.x, p0.y);
            pA[ks].u[j + 1] = pk2(p0.z, p0.w);
            pA[ks].u[j + 2] = pk2(p1.x, p1.y);
            pA[ks].u[j + 3] = pk2(p1.z, p1.w);
            cA[ks].u[j + 0] = pk2(c0.x, c0.y);
            cA[ks].u[j + 1] = pk2(c0.z, c0.w);
            cA[ks].u[j + 2] = pk2(c1.x, c1.y);
            cA[ks].u[j + 3] = pk2(c1.z, c1.w);
            dA[ks].u[j + 0] = pk2_abs(p0.x - c0.x, p0.y - c0.y);
            dA[ks].u[j + 1] = pk2_abs(p0.z - c0.z, p0.w - c0.w);
            dA[ks].u[j + 2] = pk2_abs(p1.x - c1.x, p1.y - c1.y);
            dA[ks].u[j + 3] = pk2_abs(p1.z - c1.z, p1.w - c1.w);
        }
    }
}

// Layer for two M-tiles: leaky_relu( A @ W^T + b ); each B fragment feeds 2 WMMAs.
__device__ __forceinline__ void enc_gemm2(const unsigned char* wlds, const float* biasv,
                                          const Frag* A0, const Frag* A1, unsigned short* stg,
                                          int lane, int lrow, int hl) {
    #pragma unroll
    for (int nt = 0; nt < NT; ++nt) {
        float b = biasv[nt * 16 + lrow];
        v8f a0, a1;
        #pragma unroll
        for (int r = 0; r < 8; ++r) { a0[r] = b; a1[r] = b; }
        #pragma unroll
        for (int ks = 0; ks < NK; ++ks) {
            Frag w = load_b(wlds, nt * NK + ks, lane);
            a0 = __builtin_amdgcn_wmma_f32_16x16x32_bf16(false, A0[ks].v, false, w.v,
                                                         (short)0, a0, false, false);
            a1 = __builtin_amdgcn_wmma_f32_16x16x32_bf16(false, A1[ks].v, false, w.v,
                                                         (short)0, a1, false, false);
        }
        float v0[8], v1[8];
        #pragma unroll
        for (int r = 0; r < 8; ++r) {
            v0[r] = fmaxf(a0[r], 0.01f * a0[r]);   // LeakyReLU
            v1[r] = fmaxf(a1[r], 0.01f * a1[r]);
        }
        store_acc_bf16(stg, nt * 16 + lrow, hl, v0);                     // rows 0..15
        store_acc_bf16(stg + 16 * STG_STRIDE, nt * 16 + lrow, hl, v1);   // rows 16..31
    }
}

__global__ __launch_bounds__(256, 1)
void GraphEdgeAttr_47493748359689_kernel(const float* __restrict__ x_src,
                                         const float* __restrict__ x_dst,
                                         const int*   __restrict__ ei,     // [2, E]
                                         const float* __restrict__ Wp, const float* __restrict__ bp,
                                         const float* __restrict__ Wc, const float* __restrict__ bc,
                                         const float* __restrict__ W1, const float* __restrict__ b1,
                                         const float* __restrict__ W2, const float* __restrict__ b2,
                                         float* __restrict__ out, int E) {
    extern __shared__ __align__(16) unsigned char smem[];
    unsigned char* ldsWp = smem + WP_OFF;
    unsigned char* ldsWc = smem + WC_OFF;
    unsigned char* ldsW1 = smem + W1_OFF;
    unsigned char* ldsW2 = smem + W2_OFF;
    float*         biasL = (float*)(smem + BIAS_OFF);

    // ---- Stage weights (bf16, B-fragment swizzled) + biases into LDS ----
    stage_w(ldsWp, Wp, HD, HD);
    stage_w(ldsWc, Wc, HD, HD);
    stage_w(ldsW1, W1, HD, 3 * HD);
    stage_w(ldsW2, W2, HD, HD);
    {
        const float* bsrc[4] = {bp, bc, b1, b2};
        for (int i = threadIdx.x; i < 4 * HD; i += blockDim.x)
            biasL[i] = bsrc[i >> 7][i & (HD - 1)];
    }
    __syncthreads();

    const int lane = threadIdx.x & 31;
    const int wv   = threadIdx.x >> 5;
    const int hl   = lane >> 4;       // lane half (K split)
    const int lrow = lane & 15;       // M row (A) / N col (B/C/D)
    unsigned short* stg = (unsigned short*)(smem + STG_OFF) + wv * 32 * STG_STRIDE;

    const int nBT = (E + 255) >> 8;   // 256 edges per block-tile (32 per wave)
    for (int bt = blockIdx.x; bt < nBT; bt += gridDim.x) {
        int eb = bt * 256 + wv * 32;  // wave-uniform
        if (eb >= E) continue;        // E % 32 == 0 -> full 2x16 tiles only

        // ---- Gather two 16-edge A-tiles; diff on the fly ----
        Frag pA0[NK], cA0[NK], dA0[NK], pA1[NK], cA1[NK], dA1[NK];
        gather_frags(x_src, x_dst, ei, E, eb + lrow,      hl, pA0, cA0, dA0);
        gather_frags(x_src, x_dst, ei, E, eb + 16 + lrow, hl, pA1, cA1, dA1);

        // ---- enc_p = leaky_relu(par @ Wp^T + bp) ; transpose via staging ----
        enc_gemm2(ldsWp, biasL + 0 * HD, pA0, pA1, stg, lane, lrow, hl);
        read_stage_A(stg,                   lrow, hl, pA0);
        read_stage_A(stg + 16 * STG_STRIDE, lrow, hl, pA1);

        // ---- enc_c = leaky_relu(cld @ Wc^T + bc) ----
        enc_gemm2(ldsWc, biasL + 1 * HD, cA0, cA1, stg, lane, lrow, hl);
        read_stage_A(stg,                   lrow, hl, cA0);
        read_stage_A(stg + 16 * STG_STRIDE, lrow, hl, cA1);

        // ---- H = relu([enc_p|enc_c|diff] @ W1^T + b1) : 3 K-segments of W1 ----
        #pragma unroll
        for (int nt = 0; nt < NT; ++nt) {
            float b = biasL[2 * HD + nt * 16 + lrow];
            v8f a0, a1;
            #pragma unroll
            for (int r = 0; r < 8; ++r) { a0[r] = b; a1[r] = b; }
            #pragma unroll
            for (int ks = 0; ks < NK; ++ks) {
                Frag w = load_b(ldsW1, nt * NK1 + ks, lane);
                a0 = __builtin_amdgcn_wmma_f32_16x16x32_bf16(false, pA0[ks].v, false, w.v,
                                                             (short)0, a0, false, false);
                a1 = __builtin_amdgcn_wmma_f32_16x16x32_bf16(false, pA1[ks].v, false, w.v,
                                                             (short)0, a1, false, false);
            }
            #pragma unroll
            for (int ks = 0; ks < NK; ++ks) {
                Frag w = load_b(ldsW1, nt * NK1 + 4 + ks, lane);
                a0 = __builtin_amdgcn_wmma_f32_16x16x32_bf16(false, cA0[ks].v, false, w.v,
                                                             (short)0, a0, false, false);
                a1 = __builtin_amdgcn_wmma_f32_16x16x32_bf16(false, cA1[ks].v, false, w.v,
                                                             (short)0, a1, false, false);
            }
            #pragma unroll
            for (int ks = 0; ks < NK; ++ks) {
                Frag w = load_b(ldsW1, nt * NK1 + 8 + ks, lane);
                a0 = __builtin_amdgcn_wmma_f32_16x16x32_bf16(false, dA0[ks].v, false, w.v,
                                                             (short)0, a0, false, false);
                a1 = __builtin_amdgcn_wmma_f32_16x16x32_bf16(false, dA1[ks].v, false, w.v,
                                                             (short)0, a1, false, false);
            }
            float v0[8], v1[8];
            #pragma unroll
            for (int r = 0; r < 8; ++r) {
                v0[r] = fmaxf(a0[r], 0.0f);        // ReLU
                v1[r] = fmaxf(a1[r], 0.0f);
            }
            store_acc_bf16(stg, nt * 16 + lrow, hl, v0);
            store_acc_bf16(stg + 16 * STG_STRIDE, nt * 16 + lrow, hl, v1);
        }
        read_stage_A(stg,                   lrow, hl, dA0);   // dA := H A-fragments
        read_stage_A(stg + 16 * STG_STRIDE, lrow, hl, dA1);

        // ---- out = H @ W2^T + b2 (f32 store, coalesced across lanes) ----
        #pragma unroll
        for (int nt = 0; nt < NT; ++nt) {
            float b = biasL[3 * HD + nt * 16 + lrow];
            v8f a0, a1;
            #pragma unroll
            for (int r = 0; r < 8; ++r) { a0[r] = b; a1[r] = b; }
            #pragma unroll
            for (int ks = 0; ks < NK; ++ks) {
                Frag w = load_b(ldsW2, nt * NK + ks, lane);
                a0 = __builtin_amdgcn_wmma_f32_16x16x32_bf16(false, dA0[ks].v, false, w.v,
                                                             (short)0, a0, false, false);
                a1 = __builtin_amdgcn_wmma_f32_16x16x32_bf16(false, dA1[ks].v, false, w.v,
                                                             (short)0, a1, false, false);
            }
            #pragma unroll
            for (int r = 0; r < 8; ++r) {
                out[(size_t)(eb +      r + 8 * hl) * HD + nt * 16 + lrow] = a0[r];
                out[(size_t)(eb + 16 + r + 8 * hl) * HD + nt * 16 + lrow] = a1[r];
            }
        }
    }
}

extern "C" void kernel_launch(void* const* d_in, const int* in_sizes, int n_in,
                              void* d_out, int out_size, void* d_ws, size_t ws_size,
                              hipStream_t stream) {
    const float* x_src = (const float*)d_in[0];
    const float* x_dst = (const float*)d_in[1];
    const int*   ei    = (const int*)d_in[2];
    const float* Wp = (const float*)d_in[3];
    const float* bp = (const float*)d_in[4];
    const float* Wc = (const float*)d_in[5];
    const float* bc = (const float*)d_in[6];
    const float* W1 = (const float*)d_in[7];
    const float* b1 = (const float*)d_in[8];
    const float* W2 = (const float*)d_in[9];
    const float* b2 = (const float*)d_in[10];
    float* out = (float*)d_out;

    int E   = in_sizes[2] / 2;
    int nBT = (E + 255) >> 8;
    int grid = nBT < 320 ? nBT : 320;

    (void)hipFuncSetAttribute((const void*)GraphEdgeAttr_47493748359689_kernel,
                              hipFuncAttributeMaxDynamicSharedMemorySize,
                              (int)SMEM_BYTES);

    hipLaunchKernelGGL(GraphEdgeAttr_47493748359689_kernel,
                       dim3(grid), dim3(256), SMEM_BYTES, stream,
                       x_src, x_dst, ei, Wp, bp, Wc, bc, W1, b1, W2, b2, out, E);
}
